// FastAttention_64793876627782
// MI455X (gfx1250) — matrix operand order repaired and checked by reading, not compiled, not run on hardware
//
#include <hip/hip_runtime.h>
#include <hip/hip_bf16.h>

#define DEVINL __device__ __forceinline__

typedef __attribute__((ext_vector_type(16))) __bf16 v16bf;
typedef __attribute__((ext_vector_type(8)))  float  v8f;
typedef __attribute__((ext_vector_type(4)))  unsigned int u32x4;
typedef __attribute__((ext_vector_type(8)))  int i32x8;
typedef __attribute__((ext_vector_type(4)))  int i32x4;

constexpr int Bb = 2, Ls = 2048, E = 1024, H = 16, D = 64, M = 64;
constexpr int BH = Bb * H;                 // 32
constexpr int E3 = 3 * E;                  // 3072
constexpr int CHUNK = 64, NC = Ls / CHUNK; // 32 chunks
constexpr int KGEMM = 1024;                // K of both big GEMMs

// ---------------- WMMA fragment loaders (wave32, 16x16x32 bf16) ----------------
// A (16x32): lane row = lane&15; elems 0-7 -> K = 8*half + i ; elems 8-15 -> K = 16 + 8*half + (i-8)
DEVINL v16bf fragA_f32(const float* p, int half) {
  const float* g0 = p + 8 * half;
  const float* g1 = p + 16 + 8 * half;
  v16bf r;
#pragma unroll
  for (int i = 0; i < 8; ++i) r[i] = (__bf16)g0[i];
#pragma unroll
  for (int i = 0; i < 8; ++i) r[i + 8] = (__bf16)g1[i];
  return r;
}
DEVINL v16bf fragA_bf16(const __bf16* p, int half) {
  const __bf16* g0 = p + 8 * half;
  const __bf16* g1 = p + 16 + 8 * half;
  v16bf r;
#pragma unroll
  for (int i = 0; i < 8; ++i) r[i] = g0[i];
#pragma unroll
  for (int i = 0; i < 8; ++i) r[i + 8] = g1[i];
  return r;
}
// B (32x16): lane col = lane&15; elems 0-7 -> K = 8*half + i ; elems 8-15 -> K = 16 + 8*half + (i-8)
DEVINL v16bf fragB_bf16(const __bf16* p, int half) {
  const __bf16* g0 = p + 8 * half;
  const __bf16* g1 = p + 16 + 8 * half;
  v16bf r;
#pragma unroll
  for (int i = 0; i < 8; ++i) r[i] = g0[i];
#pragma unroll
  for (int i = 0; i < 8; ++i) r[i + 8] = g1[i];
  return r;
}
DEVINL v16bf fragB_f32(const float* p, int half) {
  const float* g0 = p + 8 * half;
  const float* g1 = p + 16 + 8 * half;
  v16bf r;
#pragma unroll
  for (int i = 0; i < 8; ++i) r[i] = (__bf16)g0[i];
#pragma unroll
  for (int i = 0; i < 8; ++i) r[i + 8] = (__bf16)g1[i];
  return r;
}
DEVINL v8f wmma_bf16(v16bf a, v16bf b, v8f c) {
  return __builtin_amdgcn_wmma_f32_16x16x32_bf16(false, a, false, b, (short)0, c, false, false);
}
// C/D (16x16 f32, 8 VGPRs): elem r -> row = r + 8*half, col = lane&15

// ---------------- TDM: 2D tile (bf16) Global -> LDS ----------------
// D# per CDNA5 ISA 8.3/8.4: group0 = {count, lds_addr, global_addr, type=2},
// group1 = {data_size, tensor dims, tile dims, stride}; groups 2/3 zero (2D tensor).
// This toolchain uses the 6-arg builtin: (u32x4, i32x8, i32x4, i32x4, i32x8, cpol).
DEVINL void tdm_load_2d_bf16(unsigned int lds_addr, const void* gaddr,
                             unsigned int tensor_d0, unsigned int tensor_d1,
                             unsigned int tile_d0, unsigned int tile_d1,
                             unsigned long long stride0_elems) {
  unsigned long long ga = (unsigned long long)(uintptr_t)gaddr;
  u32x4 g0;
  g0[0] = 1u;                                                  // count=1 (valid user D#)
  g0[1] = lds_addr;                                            // bits 63:32 lds_addr
  g0[2] = (unsigned int)ga;                                    // global_addr[31:0]
  g0[3] = (unsigned int)((ga >> 32) & 0x1FFFFFFu) | (2u << 30);// addr[56:32] | type=2
  i32x8 g1;
  g1[0] = (int)(1u << 16);                                     // data_size=1 (2 bytes)
  g1[1] = (int)((tensor_d0 & 0xFFFFu) << 16);                  // tensor_dim0[15:0]
  g1[2] = (int)((tensor_d0 >> 16) | ((tensor_d1 & 0xFFFFu) << 16));
  g1[3] = (int)((tensor_d1 >> 16) | ((tile_d0 & 0xFFFFu) << 16));
  g1[4] = (int)(tile_d1 & 0xFFFFu);                            // tile_dim1, tile_dim2=0
  g1[5] = (int)(unsigned int)stride0_elems;                    // tensor_dim0_stride[31:0]
  g1[6] = (int)((unsigned int)(stride0_elems >> 32) & 0xFFFFu);
  g1[7] = 0;
  i32x4 z4 = {0, 0, 0, 0};
  i32x8 z8 = {0, 0, 0, 0, 0, 0, 0, 0};
  __builtin_amdgcn_tensor_load_to_lds(g0, g1, z4, z4, z8, 0);
}

// ---------------- K0: weight transpose + bf16 convert ----------------
__global__ void convert_weights(const float* __restrict__ qkv_w, const float* __restrict__ out_w,
                                const float* __restrict__ proj,
                                __bf16* __restrict__ qkv_wT, __bf16* __restrict__ out_wT,
                                __bf16* __restrict__ projT) {
  int stride = gridDim.x * blockDim.x;
  int tid = blockIdx.x * blockDim.x + threadIdx.x;
  for (int i = tid; i < E3 * E; i += stride) {          // qkv_wT[n][k] = qkv_w[k][n]
    int n = i >> 10, k = i & (E - 1);
    qkv_wT[i] = (__bf16)qkv_w[(size_t)k * E3 + n];
  }
  for (int i = tid; i < E * E; i += stride) {           // out_wT[n][k] = out_w[k][n]
    int n = i >> 10, k = i & (E - 1);
    out_wT[i] = (__bf16)out_w[(size_t)k * E + n];
  }
  for (int i = tid; i < H * M * D; i += stride) {       // projT[h][m][d] = proj[h][d][m]
    int h = i >> 12, m = (i >> 6) & 63, d = i & 63;
    projT[i] = (__bf16)proj[((size_t)h * D + d) * M + m];
  }
}

// ---------------- K1/K7: C = A(f32, MxK) @ BT(bf16, NxK)^T + bias ----------------
// Block: 128 rows x 32 cols, 8 waves. The 32xK bf16 weight panel (64KB) is staged
// into LDS once per block by a single TDM tensor_load_to_lds DMA; A streams from
// global with on-the-fly f32->bf16 conversion.
__global__ __launch_bounds__(256)
void gemm_f32xbf16T(const float* __restrict__ A, const __bf16* __restrict__ BT,
                    const float* __restrict__ bias, float* __restrict__ Cmat,
                    int N, int K) {
  __shared__ __bf16 Bs[32 * KGEMM];  // 64 KB panel: 32 columns x K
  int w = threadIdx.x >> 5, lane = threadIdx.x & 31;
  int half = lane >> 4, nn = lane & 15;
  int row0 = blockIdx.y * 128 + w * 16;
  int col0 = blockIdx.x * 32;

  if (w == 0) {  // wave 0 issues the DMA and waits on TENSORcnt
    tdm_load_2d_bf16((unsigned int)(uintptr_t)&Bs[0], BT + (size_t)col0 * K,
                     (unsigned int)K, 32u, (unsigned int)K, 32u,
                     (unsigned long long)K);
    __builtin_amdgcn_s_wait_tensorcnt(0);
  }
  __syncthreads();

  const float* arow = A + (size_t)(row0 + nn) * K;
  const __bf16* b0 = &Bs[(size_t)nn * K];
  const __bf16* b1 = &Bs[(size_t)(nn + 16) * K];
  v8f acc0 = {}, acc1 = {};
  for (int kk = 0; kk < K; kk += 32) {
    v16bf a = fragA_f32(arow + kk, half);
    acc0 = wmma_bf16(a, fragB_bf16(b0 + kk, half), acc0);
    acc1 = wmma_bf16(a, fragB_bf16(b1 + kk, half), acc1);
  }
  float bi0 = bias[col0 + nn], bi1 = bias[col0 + 16 + nn];
#pragma unroll
  for (int r = 0; r < 8; ++r) {
    size_t row = (size_t)(row0 + r + 8 * half);
    Cmat[row * N + col0 + nn] = acc0[r] + bi0;
    Cmat[row * N + col0 + 16 + nn] = acc1[r] + bi1;
  }
}

// ---------------- K2: q' / k' = relu( q|k @ proj_h ), emit qp, kp, kpT ----------------
// One wave per block: 16 rows(l) x 64 cols(m), K = D = 64.
__global__ __launch_bounds__(32)
void project_qk(const float* __restrict__ qkv, const __bf16* __restrict__ projT,
                __bf16* __restrict__ qp, __bf16* __restrict__ kp, __bf16* __restrict__ kpT) {
  int lane = threadIdx.x & 31;
  int half = lane >> 4, nn = lane & 15;
  int l0 = blockIdx.x * 16;
  int bh = blockIdx.y, b = bh >> 4, h = bh & 15;
  int mat = blockIdx.z;  // 0=q, 1=k
  const float* Abase = qkv + (size_t)(b * Ls + l0 + nn) * E3 + mat * E + h * D;
  v16bf a0 = fragA_f32(Abase, half);       // d = 0..31
  v16bf a1 = fragA_f32(Abase + 32, half);  // d = 32..63
  const __bf16* Pb = projT + (size_t)h * M * D;
  __bf16* outp = (mat == 0 ? qp : kp) + (size_t)bh * Ls * M;
#pragma unroll
  for (int tc = 0; tc < 4; ++tc) {
    int col0 = tc * 16;
    const __bf16* bcol = Pb + (size_t)(col0 + nn) * D;
    v8f acc = {};
    acc = wmma_bf16(a0, fragB_bf16(bcol, half), acc);
    acc = wmma_bf16(a1, fragB_bf16(bcol + 32, half), acc);
#pragma unroll
    for (int r = 0; r < 8; ++r) {
      float v = acc[r] > 0.0f ? acc[r] : 0.0f;
      int row = l0 + r + 8 * half, col = col0 + nn;
      outp[(size_t)row * M + col] = (__bf16)v;
      if (mat == 1) kpT[(size_t)bh * M * Ls + (size_t)col * Ls + row] = (__bf16)v;
    }
  }
}

// ---------------- K3: vT[bh][d][l] = v (transposed, bf16) ----------------
__global__ void make_vT(const float* __restrict__ qkv, __bf16* __restrict__ vT) {
  int idx = blockIdx.x * 256 + threadIdx.x;  // idx = (bh*64 + d)*2048 + l
  int l = idx & (Ls - 1);
  int d = (idx >> 11) & 63;
  int bh = idx >> 17;
  int b = bh >> 4, h = bh & 15;
  vT[idx] = (__bf16)qkv[(size_t)(b * Ls + l) * E3 + 2 * E + h * D + d];
}

// ---------------- K4: per-chunk KVT_c = v_c^T @ k'_c  (D x M) and ksum_c ----------------
__global__ __launch_bounds__(512)
void chunk_state(const __bf16* __restrict__ kp, const __bf16* __restrict__ kpT,
                 const __bf16* __restrict__ vT, float* __restrict__ KVT,
                 float* __restrict__ ksum) {
  int c = blockIdx.x, bh = blockIdx.y;
  int w = threadIdx.x >> 5, lane = threadIdx.x & 31;
  int half = lane >> 4, nn = lane & 15;
  int d0 = (w >> 2) * 16, m0 = (w & 3) * 16;
  const __bf16* arow = vT + (size_t)bh * D * Ls + (size_t)(d0 + nn) * Ls + c * CHUNK;
  const __bf16* bcol = kpT + (size_t)bh * M * Ls + (size_t)(m0 + nn) * Ls + c * CHUNK;
  v8f acc = {};
#pragma unroll
  for (int kk = 0; kk < CHUNK; kk += 32)
    acc = wmma_bf16(fragA_bf16(arow + kk, half), fragB_bf16(bcol + kk, half), acc);
  float* out = KVT + ((size_t)bh * NC + c) * D * M;
#pragma unroll
  for (int r = 0; r < 8; ++r)
    out[(size_t)(d0 + r + 8 * half) * M + m0 + nn] = acc[r];
  if (threadIdx.x < M) {
    int m = threadIdx.x;
    const __bf16* kc = kp + (size_t)bh * Ls * M + (size_t)c * CHUNK * M + m;
    float s = 0.0f;
    for (int j = 0; j < CHUNK; ++j) s += (float)kc[(size_t)j * M];
    ksum[((size_t)bh * NC + c) * M + m] = s;
  }
}

// ---------------- K5: exclusive prefix over chunks (in place) ----------------
__global__ __launch_bounds__(512)
void exclusive_scan(float* __restrict__ KVT, float* __restrict__ ksum) {
  int bh = blockIdx.x;
  for (int e = threadIdx.x; e < D * M; e += 512) {
    float run = 0.0f;
    float* p = KVT + (size_t)bh * NC * D * M + e;
    for (int c = 0; c < NC; ++c) { float t = p[(size_t)c * D * M]; p[(size_t)c * D * M] = run; run += t; }
  }
  if (threadIdx.x < M) {
    float run = 0.0f;
    float* p = ksum + (size_t)bh * NC * M + threadIdx.x;
    for (int c = 0; c < NC; ++c) { float t = p[c * M]; p[c * M] = run; run += t; }
  }
}

// ---------------- K6: chunk attention ----------------
// ctx = q'_c @ S_prev + tril(q'_c @ k'_c^T) @ v_c ; norm = q'.ksum_prev + rowsum(masked QK)
__global__ __launch_bounds__(512)
void attn_chunk(const __bf16* __restrict__ qp, const __bf16* __restrict__ kp,
                const __bf16* __restrict__ vT, const float* __restrict__ STprev,
                const float* __restrict__ ksumprev, float* __restrict__ attn) {
  __shared__ float qkf[CHUNK * CHUNK];
  __shared__ __bf16 qkb[CHUNK * CHUNK];
  __shared__ float normv[CHUNK];
  int c = blockIdx.x, bh = blockIdx.y;
  int b = bh >> 4, h = bh & 15;
  int w = threadIdx.x >> 5, lane = threadIdx.x & 31;
  int half = lane >> 4, nn = lane & 15;
  int tr = w >> 2, tc = w & 3;
  const __bf16* qpc = qp + (size_t)bh * Ls * M + (size_t)c * CHUNK * M;
  const __bf16* kpc = kp + (size_t)bh * Ls * M + (size_t)c * CHUNK * M;
  const __bf16* vTb = vT + (size_t)bh * D * Ls;
  const float* STc = STprev + ((size_t)bh * NC + c) * D * M;  // [d][m]
  const float* ksc = ksumprev + ((size_t)bh * NC + c) * M;

  // phase 1: masked QK tile (16x16 each, 16 waves cover 64x64)
  {
    int i0 = tr * 16, j0 = tc * 16;
    const __bf16* arow = qpc + (size_t)(i0 + nn) * M;
    const __bf16* bcol = kpc + (size_t)(j0 + nn) * M;
    v8f acc = {};
#pragma unroll
    for (int kk = 0; kk < M; kk += 32)
      acc = wmma_bf16(fragA_bf16(arow + kk, half), fragB_bf16(bcol + kk, half), acc);
#pragma unroll
    for (int r = 0; r < 8; ++r) {
      int i = i0 + r + 8 * half, j = j0 + nn;
      float v = (j <= i) ? acc[r] : 0.0f;  // inclusive causal mask
      qkf[i * CHUNK + j] = v;
      qkb[i * CHUNK + j] = (__bf16)v;
    }
  }
  __syncthreads();

  // phase 2: context tile (rows i0, cols d0)
  v8f ctx = {};
  {
    int i0 = tr * 16, d0 = tc * 16;
    const __bf16* arow = qpc + (size_t)(i0 + nn) * M;
    const float* stcol = STc + (size_t)(d0 + nn) * M;  // B-side: fixed d, contiguous m
#pragma unroll
    for (int kk = 0; kk < M; kk += 32)  // inter-chunk: q' @ S_prev
      ctx = wmma_bf16(fragA_bf16(arow + kk, half), fragB_f32(stcol + kk, half), ctx);
    const __bf16* qrow = &qkb[(i0 + nn) * CHUNK];
    const __bf16* vcol = vTb + (size_t)(d0 + nn) * Ls + (size_t)c * CHUNK;
#pragma unroll
    for (int kk = 0; kk < CHUNK; kk += 32)  // intra-chunk: masked QK @ v
      ctx = wmma_bf16(fragA_bf16(qrow + kk, half), fragB_bf16(vcol + kk, half), ctx);
  }

  // norm (threads 0..63)
  if (threadIdx.x < CHUNK) {
    int i = threadIdx.x;
    float s = 0.0f;
#pragma unroll 8
    for (int j = 0; j < CHUNK; ++j) s += qkf[i * CHUNK + j];
    const __bf16* qrow = qpc + (size_t)i * M;
#pragma unroll 8
    for (int m = 0; m < M; ++m) s += (float)qrow[m] * ksc[m];
    normv[i] = s + 1e-6f;
  }
  __syncthreads();

  // store: attn[b][l][h*64 + d] = ctx / norm
  {
    int i0 = tr * 16, d0 = tc * 16;
#pragma unroll
    for (int r = 0; r < 8; ++r) {
      int i = i0 + r + 8 * half;
      int l = c * CHUNK + i;
      attn[(size_t)(b * Ls + l) * E + h * D + d0 + nn] = ctx[r] / normv[i];
    }
  }
}

// ---------------- launcher ----------------
extern "C" void kernel_launch(void* const* d_in, const int* in_sizes, int n_in,
                              void* d_out, int out_size, void* d_ws, size_t ws_size,
                              hipStream_t stream) {
  (void)in_sizes; (void)n_in; (void)out_size; (void)ws_size;
  const float* x      = (const float*)d_in[0];
  const float* qkv_w  = (const float*)d_in[1];
  const float* qkv_b  = (const float*)d_in[2];
  const float* out_w  = (const float*)d_in[3];
  const float* out_b  = (const float*)d_in[4];
  const float* proj   = (const float*)d_in[5];
  float* out = (float*)d_out;

  char* ws = (char*)d_ws;
  size_t o = 0;
  auto alloc = [&](size_t bytes) { void* p = ws + o; o += (bytes + 255) & ~(size_t)255; return p; };
  float*  qkv    = (float*)alloc((size_t)Bb * Ls * E3 * 4);
  float*  attn   = (float*)alloc((size_t)Bb * Ls * E * 4);
  float*  KVT    = (float*)alloc((size_t)BH * NC * D * M * 4);
  float*  ksum   = (float*)alloc((size_t)BH * NC * M * 4);
  __bf16* qp     = (__bf16*)alloc((size_t)BH * Ls * M * 2);
  __bf16* kp     = (__bf16*)alloc((size_t)BH * Ls * M * 2);
  __bf16* kpT    = (__bf16*)alloc((size_t)BH * Ls * M * 2);
  __bf16* vT     = (__bf16*)alloc((size_t)BH * D * Ls * 2);
  __bf16* qkv_wT = (__bf16*)alloc((size_t)E3 * E * 2);
  __bf16* out_wT = (__bf16*)alloc((size_t)E * E * 2);
  __bf16* projT  = (__bf16*)alloc((size_t)H * M * D * 2);

  convert_weights<<<2048, 256, 0, stream>>>(qkv_w, out_w, proj, qkv_wT, out_wT, projT);
  gemm_f32xbf16T<<<dim3(E3 / 32, (Bb * Ls) / 128), 256, 0, stream>>>(x, qkv_wT, qkv_b, qkv, E3, KGEMM);
  project_qk<<<dim3(Ls / 16, BH, 2), 32, 0, stream>>>(qkv, projT, qp, kp, kpT);
  make_vT<<<(BH * D * Ls) / 256, 256, 0, stream>>>(qkv, vT);
  chunk_state<<<dim3(NC, BH), 512, 0, stream>>>(kp, kpT, vT, KVT, ksum);
  exclusive_scan<<<BH, 512, 0, stream>>>(KVT, ksum);
  attn_chunk<<<dim3(NC, BH), 512, 0, stream>>>(qp, kp, vT, KVT, ksum, attn);
  gemm_f32xbf16T<<<dim3(E / 32, (Bb * Ls) / 128), 256, 0, stream>>>(attn, out_wT, out_b, out, E, KGEMM);
}
